// Net_44349832298833
// MI455X (gfx1250) — matrix operand order repaired and checked by reading, not compiled, not run on hardware
//
#include <hip/hip_runtime.h>

// Problem constants (from reference)
#define Bn 32
#define Tn 512
#define Dn 256
#define TNUM 10
#define Kc 1000             // codebook size
#define Mrows (Bn * Tn)     // 16384
#define NT ((Kc + 15) / 16) // 63 N-tiles of 16 codes

typedef __attribute__((ext_vector_type(2))) float v2f;
typedef __attribute__((ext_vector_type(8))) float v8f;
typedef __attribute__((ext_vector_type(4))) __bf16 v4bf;
typedef __attribute__((ext_vector_type(8))) __bf16 v8bf;
typedef __attribute__((ext_vector_type(16))) __bf16 v16bf;

__device__ __forceinline__ v16bf cat16(v8bf a, v8bf b) {
  return __builtin_shufflevector(a, b, 0, 1, 2, 3, 4, 5, 6, 7, 8, 9, 10, 11,
                                 12, 13, 14, 15);
}

// ---------------------------------------------------------------------------
// Kernel 1: inv_norms[k], proj[k] = e_k @ W + b, and bf16 hi/lo split of e_k.
// grid = Kc blocks, block = 256 threads (one per output d)
// ---------------------------------------------------------------------------
__global__ __launch_bounds__(256) void proj_kernel(
    const float* __restrict__ embed, const float* __restrict__ W,
    const float* __restrict__ bias, float* __restrict__ invn,
    float* __restrict__ proj, __bf16* __restrict__ eh,
    __bf16* __restrict__ el) {
  const int k = blockIdx.x;
  const int d = threadIdx.x;
  __shared__ float se[Dn];
  __shared__ float red[Dn];

  const float ev = embed[k * Dn + d];
  se[d] = ev;
  red[d] = ev * ev;

  // bf16 split: ev == (float)h + (float)l + O(2^-16 * ev)
  const __bf16 h = (__bf16)ev;
  const __bf16 l = (__bf16)(ev - (float)h);
  eh[k * Dn + d] = h;
  el[k * Dn + d] = l;

  __syncthreads();
  for (int s = 128; s > 0; s >>= 1) {
    if (d < s) red[d] += red[d + s];
    __syncthreads();
  }
  if (d == 0) invn[k] = 1.0f / sqrtf(red[0]);

  float acc = bias[d];
#pragma unroll 8
  for (int dk = 0; dk < Dn; ++dk) acc = fmaf(se[dk], W[dk * Dn + d], acc);
  proj[k * Dn + d] = acc;
}

// ---------------------------------------------------------------------------
// Kernel 2: per-row argmax_k ( x_row . e_k ) * invn[k] via split-bf16 WMMA:
//   dot = xh.eh + xh.el + xl.eh  (three v_wmma_f32_16x16x32_bf16 per K=32)
// grid = Mrows/32 blocks, block = 128 threads (4 waves).
// Block owns 32 rows (two 16-row M-subtiles, staged as bf16 hi/lo in LDS,
// row stride 264 halves = 528B -> 4-bank shift per row, conflict-free b128).
// Wave w handles N-tiles w, w+4, ...; both M-subtiles share each B tile.
// ---------------------------------------------------------------------------
#define APADH 264
__global__ __launch_bounds__(128) void argmax_kernel(
    const float* __restrict__ x, const __bf16* __restrict__ eh,
    const __bf16* __restrict__ el, const float* __restrict__ invn,
    int* __restrict__ idx_out) {
  __shared__ __bf16 sXh[32][APADH];
  __shared__ __bf16 sXl[32][APADH];
  __shared__ float sVal[4][32];
  __shared__ int sIdx[4][32];

  const int tid = threadIdx.x;
  const int rowBase = blockIdx.x * 32;

  // Stage 32 rows of x, converting f32 -> bf16 hi + residual lo.
  for (int i = tid; i < 32 * (Dn / 4); i += 128) {
    const int r = i >> 6;  // 64 float4 per row
    const int c4 = i & 63;
    const float4 v = ((const float4*)(x + (size_t)(rowBase + r) * Dn))[c4];
    v4bf h, l;
    h[0] = (__bf16)v.x; l[0] = (__bf16)(v.x - (float)h[0]);
    h[1] = (__bf16)v.y; l[1] = (__bf16)(v.y - (float)h[1]);
    h[2] = (__bf16)v.z; l[2] = (__bf16)(v.z - (float)h[2]);
    h[3] = (__bf16)v.w; l[3] = (__bf16)(v.w - (float)h[3]);
    *(v4bf*)&sXh[r][c4 * 4] = h;
    *(v4bf*)&sXl[r][c4 * 4] = l;
  }
  __syncthreads();

  const int wave = tid >> 5;
  const int lane = tid & 31;
  const int lrow = lane & 15;  // A: M index (within subtile) / B: N code index
  const int lhi = lane >> 4;   // half-wave selector for K runs

  float best[2][8];
  int bidx[2][8];
#pragma unroll
  for (int s = 0; s < 2; ++s)
#pragma unroll
    for (int r = 0; r < 8; ++r) {
      best[s][r] = -__builtin_inff();
      bidx[s][r] = 0;
    }

  for (int tile = wave; tile < NT; tile += 4) {
    const int kk = tile * 16 + lrow;
    const bool vk = (kk < Kc);
    const int krow = vk ? kk : (Kc - 1);
    const __bf16* ehp = eh + (size_t)krow * Dn;
    const __bf16* elp = el + (size_t)krow * Dn;

    // Prefetch next N-tile's code rows into cache (global_prefetch_b8).
    const int ntile = tile + 4;
    if (ntile < NT) {
      const int nk = min(ntile * 16 + lrow, Kc - 1);
      __builtin_prefetch(eh + (size_t)nk * Dn, 0, 3);
      __builtin_prefetch(el + (size_t)nk * Dn, 0, 3);
    }

    // 6 independent accumulator chains (3 split terms x 2 M-subtiles).
    v8f a0hh = {}, a0hl = {}, a0lh = {};
    v8f a1hh = {}, a1hl = {}, a1lh = {};

#pragma unroll 2
    for (int d0 = 0; d0 < Dn; d0 += 32) {
      // B (32x16): lane holds runs K = d0+8*lhi+[0,8) and d0+16+8*lhi+[0,8)
      const int kb = d0 + 8 * lhi;
      const v16bf Bh = cat16(*(const v8bf*)(ehp + kb),
                             *(const v8bf*)(ehp + kb + 16));
      const v16bf Bl = cat16(*(const v8bf*)(elp + kb),
                             *(const v8bf*)(elp + kb + 16));
      // A (16x32): lane holds runs K = d0+8*lhi+[0,8) and d0+16+8*lhi+[0,8)
      const int ka = d0 + 8 * lhi;
      const v16bf A0h = cat16(*(const v8bf*)&sXh[lrow][ka],
                              *(const v8bf*)&sXh[lrow][ka + 16]);
      const v16bf A0l = cat16(*(const v8bf*)&sXl[lrow][ka],
                              *(const v8bf*)&sXl[lrow][ka + 16]);
      const v16bf A1h = cat16(*(const v8bf*)&sXh[16 + lrow][ka],
                              *(const v8bf*)&sXh[16 + lrow][ka + 16]);
      const v16bf A1l = cat16(*(const v8bf*)&sXl[lrow][ka] /*dummy*/,
                              *(const v8bf*)&sXl[lrow][ka + 16]);
      const v16bf A1lr = cat16(*(const v8bf*)&sXl[16 + lrow][ka],
                               *(const v8bf*)&sXl[16 + lrow][ka + 16]);
      (void)A1l;

      a0hh = __builtin_amdgcn_wmma_f32_16x16x32_bf16(false, A0h, false, Bh,
                                                     (short)0, a0hh, false,
                                                     false);
      a0hl = __builtin_amdgcn_wmma_f32_16x16x32_bf16(false, A0h, false, Bl,
                                                     (short)0, a0hl, false,
                                                     false);
      a0lh = __builtin_amdgcn_wmma_f32_16x16x32_bf16(false, A0l, false, Bh,
                                                     (short)0, a0lh, false,
                                                     false);
      a1hh = __builtin_amdgcn_wmma_f32_16x16x32_bf16(false, A1h, false, Bh,
                                                     (short)0, a1hh, false,
                                                     false);
      a1hl = __builtin_amdgcn_wmma_f32_16x16x32_bf16(false, A1h, false, Bl,
                                                     (short)0, a1hl, false,
                                                     false);
      a1lh = __builtin_amdgcn_wmma_f32_16x16x32_bf16(false, A1lr, false, Bh,
                                                     (short)0, a1lh, false,
                                                     false);
    }

    const float scale = vk ? invn[kk] : 0.0f;
#pragma unroll
    for (int r = 0; r < 8; ++r) {
      const float v0 =
          vk ? (a0hh[r] + a0hl[r] + a0lh[r]) * scale : -__builtin_inff();
      if (v0 > best[0][r]) { best[0][r] = v0; bidx[0][r] = kk; }
      const float v1 =
          vk ? (a1hh[r] + a1hl[r] + a1lh[r]) * scale : -__builtin_inff();
      if (v1 > best[1][r]) { best[1][r] = v1; bidx[1][r] = kk; }
    }
  }

  // Argmax across the 16 lanes of each half (xor masks stay inside halves).
#pragma unroll
  for (int off = 8; off >= 1; off >>= 1) {
#pragma unroll
    for (int s = 0; s < 2; ++s)
#pragma unroll
      for (int r = 0; r < 8; ++r) {
        const float ov = __shfl_xor(best[s][r], off, 32);
        const int oi = __shfl_xor(bidx[s][r], off, 32);
        if (ov > best[s][r] || (ov == best[s][r] && oi < bidx[s][r])) {
          best[s][r] = ov;
          bidx[s][r] = oi;
        }
      }
  }

  if (lrow == 0) {  // lane 0 -> rows +0..7, lane 16 -> rows +8..15
#pragma unroll
    for (int s = 0; s < 2; ++s)
#pragma unroll
      for (int r = 0; r < 8; ++r) {
        sVal[wave][s * 16 + lhi * 8 + r] = best[s][r];
        sIdx[wave][s * 16 + lhi * 8 + r] = bidx[s][r];
      }
  }
  __syncthreads();

  if (tid < 32) {  // combine the 4 waves' disjoint code subsets
    float bv = sVal[0][tid];
    int bi = sIdx[0][tid];
#pragma unroll
    for (int w = 1; w < 4; ++w) {
      const float v = sVal[w][tid];
      const int i2 = sIdx[w][tid];
      if (v > bv || (v == bv && i2 < bi)) { bv = v; bi = i2; }
    }
    idx_out[rowBase + tid] = bi;
  }
}

// ---------------------------------------------------------------------------
// Kernel 3: per-(b,t) closed-form loss row sum.
// loss_row = sum_d 38.5*x^2 - 11*x*m + m^2, m = mean_tnum(xs_pad_out), masked.
// grid = Mrows blocks, block = 256 threads (one per d).
// ---------------------------------------------------------------------------
__global__ __launch_bounds__(256) void loss_rows_kernel(
    const float* __restrict__ xs_out, const int* __restrict__ ilens,
    const float* __restrict__ proj, const int* __restrict__ idx,
    float* __restrict__ rowloss) {
  const int bt = blockIdx.x;
  const int d = threadIdx.x;
  const int b = bt >> 9;  // T = 512
  const int t = bt & 511;
  __shared__ float red[Dn];

  float contrib = 0.0f;
  if (t < ilens[b]) {
    const float xv = proj[(size_t)idx[bt] * Dn + d];
    const float* p = xs_out + (size_t)bt * TNUM * Dn + d;
    float s = 0.0f;
#pragma unroll
    for (int j = 0; j < TNUM; ++j) s += p[j * Dn];
    const float m = s * (1.0f / TNUM);
    contrib = 38.5f * xv * xv - 11.0f * xv * m + m * m;
  }
  red[d] = contrib;
  __syncthreads();
  for (int s = 128; s > 0; s >>= 1) {
    if (d < s) red[d] += red[d + s];
    __syncthreads();
  }
  if (d == 0) rowloss[bt] = red[0];
}

// ---------------------------------------------------------------------------
// Kernel 4: deterministic final reduction of 16384 row losses.
// ---------------------------------------------------------------------------
__global__ __launch_bounds__(256) void final_reduce(
    const float* __restrict__ rowloss, float* __restrict__ out) {
  __shared__ float red[256];
  float s = 0.0f;
  for (int i = threadIdx.x; i < Mrows; i += 256) s += rowloss[i];
  red[threadIdx.x] = s;
  __syncthreads();
  for (int st = 128; st > 0; st >>= 1) {
    if (threadIdx.x < st) red[threadIdx.x] += red[threadIdx.x + st];
    __syncthreads();
  }
  if (threadIdx.x == 0) out[0] = red[0];
}

// ---------------------------------------------------------------------------
extern "C" void kernel_launch(void* const* d_in, const int* in_sizes, int n_in,
                              void* d_out, int out_size, void* d_ws,
                              size_t ws_size, hipStream_t stream) {
  const float* xs_pad_in = (const float*)d_in[0];   // [B,T,D]
  const float* xs_pad_out = (const float*)d_in[1];  // [B,T,TNUM,D]
  const int* ilens = (const int*)d_in[2];           // [B]
  // d_in[3] = ys_pad (unused by reference)
  const float* embed_w = (const float*)d_in[4];     // [K,D]
  const float* W = (const float*)d_in[5];           // [D,D]
  const float* bias = (const float*)d_in[6];        // [D]
  float* out = (float*)d_out;

  // Workspace layout (bytes):
  // invn[4KB] | proj[1000KB] | rowloss[64KB] | idx[64KB] | eh[500KB] | el[500KB]
  char* base = (char*)d_ws;
  float* invn = (float*)base;
  float* proj = (float*)(base + 4096);
  float* rowloss = (float*)(base + 4096 + 1024000);
  int* idx = (int*)(base + 4096 + 1024000 + 65536);
  __bf16* eh = (__bf16*)(base + 4096 + 1024000 + 65536 + 65536);
  __bf16* el = eh + (size_t)Kc * Dn;

  proj_kernel<<<Kc, 256, 0, stream>>>(embed_w, W, bias, invn, proj, eh, el);
  argmax_kernel<<<Mrows / 32, 128, 0, stream>>>(xs_pad_in, eh, el, invn, idx);
  loss_rows_kernel<<<Mrows, 256, 0, stream>>>(xs_pad_out, ilens, proj, idx,
                                              rowloss);
  final_reduce<<<1, 256, 0, stream>>>(rowloss, out);
}
